// ConditionalAttentionDense_20641612824791
// MI455X (gfx1250) — matrix operand order repaired and checked by reading, not compiled, not run on hardware
//
#include <hip/hip_runtime.h>
#include <hip/hip_bf16.h>

typedef __attribute__((ext_vector_type(16))) __bf16 v16bf;
typedef __attribute__((ext_vector_type(8)))  __bf16 v8bf;
typedef __attribute__((ext_vector_type(8)))  float  v8f;
typedef __attribute__((ext_vector_type(4)))  unsigned int v4u;

#define NBATCH 16
#define NTOK   2048            // H*W
#define CCH    512
#define MTOT   (NBATCH * NTOK) // 32768

// workspace layout (bytes), all 256B aligned
#define QKT_OFF  0u                       // [128][512] bf16  (WqT|WkT)
#define WVT_OFF  131072u                  // [512][512] bf16  (WvT)
#define Q_OFF    655360u                  // [MTOT][64] bf16
#define K_OFF    4849664u                 // [MTOT][64] bf16
#define VT_OFF   9043968u                 // [NBATCH][512][2048] bf16

static __device__ __forceinline__ v8f wmma_bf16(v16bf a, v16bf b, v8f c) {
  return __builtin_amdgcn_wmma_f32_16x16x32_bf16(false, a, false, b,
                                                 (short)0, c, false, false);
}

// CDNA5 async global->LDS copy, 16B per lane, tracked by ASYNCcnt.
static __device__ __forceinline__ void async_ld_b128(unsigned lds_off, const void* gaddr) {
  asm volatile("global_load_async_to_lds_b128 %0, %1, off"
               :: "v"(lds_off), "v"(gaddr) : "memory");
}
static __device__ __forceinline__ void wait_async0() {
  asm volatile("s_wait_asynccnt 0x0" ::: "memory");
}

// A-operand fragment (16x32 bf16) from row-major [rows][ld] at (row0,k0)=base.
// ISA layout: lane<16 holds row=lane, K={0..7,16..23}; lane>=16: K={8..15,24..31}.
static __device__ __forceinline__ v16bf load_a_frag(const __bf16* base, int ld, int lane) {
  int row = lane & 15, hi = lane >> 4;
  const __bf16* p = base + (size_t)row * ld + hi * 8;
  v8bf c0 = *(const v8bf*)(p);
  v8bf c1 = *(const v8bf*)(p + 16);
  v16bf r;
  #pragma unroll
  for (int j = 0; j < 8; ++j) { r[j] = c0[j]; r[j + 8] = c1[j]; }
  return r;
}

// Same fragment but converting from an fp32 row-major source.
static __device__ __forceinline__ v16bf load_a_frag_f32(const float* base, int ld, int lane) {
  int row = lane & 15, hi = lane >> 4;
  const float* p = base + (size_t)row * ld + hi * 8;
  v8f c0 = *(const v8f*)(p);
  v8f c1 = *(const v8f*)(p + 16);
  v16bf r;
  #pragma unroll
  for (int j = 0; j < 8; ++j) { r[j] = (__bf16)c0[j]; r[j + 8] = (__bf16)c1[j]; }
  return r;
}

// B-operand fragment (32x16 bf16) from a K-contiguous row-major [Ncols][ld]
// buffer. Lane<16: col=lane, K={0..7,16..23}; lane>=16: K={8..15,24..31}.
static __device__ __forceinline__ v16bf load_b_frag(const __bf16* base, int ld, int lane) {
  int n = lane & 15, hi = lane >> 4;
  const __bf16* p = base + (size_t)n * ld + hi * 8;
  v8bf c0 = *(const v8bf*)(p);
  v8bf c1 = *(const v8bf*)(p + 16);
  v16bf r;
  #pragma unroll
  for (int j = 0; j < 8; ++j) { r[j] = c0[j]; r[j + 8] = c1[j]; }
  return r;
}

// ---------------- Kernel 1: weight convert + transpose to bf16 ----------------
__global__ void wconv_kernel(const float* __restrict__ Wq, const float* __restrict__ Wk,
                             const float* __restrict__ Wv,
                             __bf16* __restrict__ qkT, __bf16* __restrict__ wvT) {
  int i = blockIdx.x * blockDim.x + threadIdx.x;
  const int t1 = 128 * 512;
  const int t2 = 512 * 512;
  if (i < t1) {                       // qkT[n][k] = (n<64 ? Wq : Wk)[k][n%64]
    int n = i >> 9, kk = i & 511;
    float v = (n < 64) ? Wq[kk * 64 + n] : Wk[kk * 64 + (n - 64)];
    qkT[i] = (__bf16)v;
  } else if (i < t1 + t2) {           // wvT[n][k] = Wv[k][n]
    int j = i - t1;
    int n = j >> 9, kk = j & 511;
    wvT[j] = (__bf16)Wv[kk * 512 + n];
  }
}

// -------- Kernels 2/3 common core: 128-token x 128-col GEMM block with ---------
// -------- double-buffered LDS weight slab + A-fragment register prefetch -------
// wT points at the 128-column weight slice ([128][512] bf16, K-contiguous).
// Computes acc[t] for this wave's 16 tokens x (t*16..t*16+15) cols.
static __device__ __forceinline__ void gemm_block_128(
    const float* __restrict__ Xb, const __bf16* __restrict__ wT,
    __bf16* __restrict__ sw /* [2][128*32] */, int tid, int lane, v8f acc[8]) {
  int c0 = tid, c1 = tid + 256;       // 512 x 16B chunks per 8KB slab
  // prologue: stage k-step 0 + first A fragment
  v4u w0 = *(const v4u*)(wT + (size_t)(c0 >> 2) * 512 + (c0 & 3) * 8);
  v4u w1 = *(const v4u*)(wT + (size_t)(c1 >> 2) * 512 + (c1 & 3) * 8);
  *(v4u*)(sw + (c0 >> 2) * 32 + (c0 & 3) * 8) = w0;
  *(v4u*)(sw + (c1 >> 2) * 32 + (c1 & 3) * 8) = w1;
  v16bf a_cur = load_a_frag_f32(Xb, CCH, lane);
  __syncthreads();
  int cur = 0;
  for (int kk = 0; kk < CCH; kk += 32) {
    bool hn = (kk + 32) < CCH;
    v4u nw0, nw1; v16bf a_next;
    if (hn) {
      nw0 = *(const v4u*)(wT + (size_t)(c0 >> 2) * 512 + (kk + 32) + (c0 & 3) * 8);
      nw1 = *(const v4u*)(wT + (size_t)(c1 >> 2) * 512 + (kk + 32) + (c1 & 3) * 8);
      a_next = load_a_frag_f32(Xb + (kk + 32), CCH, lane);
    }
    const __bf16* swc = sw + cur * (128 * 32);
    #pragma unroll
    for (int t = 0; t < 8; ++t) {
      v16bf b = load_b_frag(swc + (t * 16) * 32, 32, lane);
      acc[t] = wmma_bf16(a_cur, b, acc[t]);
    }
    if (hn) {
      __bf16* swn = sw + (cur ^ 1) * (128 * 32);
      *(v4u*)(swn + (c0 >> 2) * 32 + (c0 & 3) * 8) = nw0;
      *(v4u*)(swn + (c1 >> 2) * 32 + (c1 & 3) * 8) = nw1;
      a_cur = a_next;
    }
    __syncthreads();
    cur ^= 1;
  }
}

// ---------------- Kernel 2: q/k projection (GEMM 32768x128x512) ----------------
// Block = 8 waves = 8 consecutive 16-token tiles; all waves share the qkT slab.
__global__ __launch_bounds__(256) void proj_qk_kernel(const float* __restrict__ X,
    const __bf16* __restrict__ qkT, const float* __restrict__ bq,
    const float* __restrict__ bk, __bf16* __restrict__ q, __bf16* __restrict__ k) {
  __shared__ __bf16 sw[2][128 * 32];
  int tid = threadIdx.x, lane = tid & 31, widx = tid >> 5;
  int tt = blockIdx.x * 8 + widx;
  const float* Xb = X + (size_t)tt * 16 * CCH;
  v8f acc[8];
  #pragma unroll
  for (int t = 0; t < 8; ++t) acc[t] = (v8f)0.0f;
  gemm_block_128(Xb, qkT, &sw[0][0], tid, lane, acc);
  int hi = lane >> 4, col0 = lane & 15;
  #pragma unroll
  for (int t = 0; t < 8; ++t) {
    int col = t * 16 + col0;
    float bias = (col < 64) ? bq[col] : bk[col - 64];
    #pragma unroll
    for (int r = 0; r < 8; ++r) {
      size_t tok = (size_t)tt * 16 + r + 8 * hi;
      float val = acc[t][r] + bias;
      if (col < 64) q[tok * 64 + col] = (__bf16)val;
      else          k[tok * 64 + (col - 64)] = (__bf16)val;
    }
  }
}

// -------- Kernel 3: v projection, output transposed vT[b][c][token] (bf16) --------
// Block = 8 waves, same 128-col slice (shared WvT slab), 8 token tiles.
__global__ __launch_bounds__(256) void proj_v_kernel(const float* __restrict__ X,
    const __bf16* __restrict__ wvT, const float* __restrict__ bv,
    __bf16* __restrict__ vT) {
  __shared__ __bf16 sw[2][128 * 32];
  int tid = threadIdx.x, lane = tid & 31, widx = tid >> 5;
  int cs = blockIdx.x & 3;
  int tt = (blockIdx.x >> 2) * 8 + widx;
  const float* Xb = X + (size_t)tt * 16 * CCH;
  v8f acc[8];
  #pragma unroll
  for (int t = 0; t < 8; ++t) acc[t] = (v8f)0.0f;
  gemm_block_128(Xb, wvT + (size_t)(cs * 128) * 512, &sw[0][0], tid, lane, acc);
  int hi = lane >> 4, col0 = lane & 15;
  int gtok0 = tt * 16 + 8 * hi;                     // 8 consecutive tokens
  int b = gtok0 >> 11, tok = gtok0 & (NTOK - 1);
  #pragma unroll
  for (int t = 0; t < 8; ++t) {
    int c = cs * 128 + t * 16 + col0;
    float bias = bv[c];
    v8bf pk;
    #pragma unroll
    for (int r = 0; r < 8; ++r) pk[r] = (__bf16)(acc[t][r] + bias);
    *(v8bf*)(vT + (size_t)b * CCH * NTOK + (size_t)c * NTOK + tok) = pk;
  }
}

// ------- Kernel 4: flash attention, async-LDS staged, double-buffered ----------
// Workgroup = 8 waves, same (batch, channel-slice), 8 consecutive 16-query tiles.
// Per 64-key block: k (8KB) + v^T slab (16KB) staged via global_load_async_to_lds,
// drained with s_wait_asynccnt right before the block barrier.
__global__ __launch_bounds__(256) void attn_kernel(const float* __restrict__ Xin,
    const __bf16* __restrict__ q, const __bf16* __restrict__ k,
    const __bf16* __restrict__ vT, const float* __restrict__ beta_p,
    float* __restrict__ out) {
  __shared__ __bf16 sk[2][64 * 64];     // [buf][key][dqk]
  __shared__ __bf16 sv[2][128 * 64];    // [buf][c_local][tok_local]

  int tid  = threadIdx.x;
  int lane = tid & 31;
  int widx = tid >> 5;
  int blk = blockIdx.x;
  int qg = blk & 15;                 // query group: 8 tiles of 16
  int cs = (blk >> 4) & 3;           // 128-channel slice
  int b  = blk >> 6;                 // batch
  int qt = qg * 8 + widx;            // this wave's 16-query tile

  const __bf16* qb = q  + ((size_t)b * NTOK + qt * 16) * 64;
  const __bf16* kb = k  + (size_t)b * NTOK * 64;
  const __bf16* vb = vT + (size_t)b * CCH * NTOK + (size_t)(cs * 128) * NTOK;

  // issue the 6 async 16B/lane copies staging block kt into buffer buf
  auto stage_async = [&](int buf, int kt) {
    unsigned skb = (unsigned)(size_t)&sk[buf][0];
    unsigned svb = (unsigned)(size_t)&sv[buf][0];
    const __bf16* gk = kb + (size_t)kt * 64;        // contiguous 8KB
    async_ld_b128(skb + tid * 16, gk + tid * 8);
    async_ld_b128(skb + (tid + 256) * 16, gk + (tid + 256) * 8);
    #pragma unroll
    for (int i = 0; i < 4; ++i) {
      int ch = tid + i * 256;                       // 1024 x 16B chunks
      int row = ch >> 3, c16 = ch & 7;
      async_ld_b128(svb + (unsigned)(row * 64 + c16 * 8) * 2,
                    vb + (size_t)row * NTOK + kt + c16 * 8);
    }
  };

  // q tile as B operand (s^T = K * Q^T), dqk=64 -> 2 resident fragments
  v16bf qB0 = load_b_frag(qb, 64, lane);
  v16bf qB1 = load_b_frag(qb + 32, 64, lane);

  v8f acc[8];
  #pragma unroll
  for (int t = 0; t < 8; ++t) acc[t] = (v8f)0.0f;
  float m = -3.402823466e38f, lsum = 0.0f;
  const float L2E = 1.44269504088896340736f;
  int hi8 = (lane >> 4) << 3;

  stage_async(0, 0);
  wait_async0();
  __syncthreads();

  int cur = 0;
  for (int kt = 0; kt < NTOK; kt += 64) {
    bool has_next = (kt + 64) < NTOK;
    if (has_next) stage_async(cur ^ 1, kt + 64);    // fills while we compute

    // ---- s^T tiles from LDS k + resident q ----
    const __bf16* skc = sk[cur];
    const __bf16* svc = sv[cur];
    v8f sT[4];
    #pragma unroll
    for (int t = 0; t < 4; ++t) {
      const __bf16* kp = skc + (t * 16) * 64;
      v16bf a0 = load_a_frag(kp, 64, lane);
      v16bf a1 = load_a_frag(kp + 32, 64, lane);
      sT[t] = wmma_bf16(a1, qB1, wmma_bf16(a0, qB0, (v8f)0.0f));
    }
    // ---- online softmax: per-lane partial over 32 keys + one xor-16 combine ----
    float bm = sT[0][0];
    #pragma unroll
    for (int t = 0; t < 4; ++t)
      #pragma unroll
      for (int r = 0; r < 8; ++r) bm = fmaxf(bm, sT[t][r]);
    bm = fmaxf(bm, __shfl_xor(bm, 16, 32));
    float mn = fmaxf(m, bm);
    float alpha = __builtin_exp2f((m - mn) * L2E);
    float bs = 0.0f;
    v16bf pA[2];   // p in A-layout: pure per-lane repack of s^T C-layout
    #pragma unroll
    for (int t = 0; t < 4; ++t) {
      #pragma unroll
      for (int r = 0; r < 8; ++r) {
        float e = __builtin_exp2f((sT[t][r] - mn) * L2E);
        bs += e;
        pA[t >> 1][(t & 1) * 8 + r] = (__bf16)e;
      }
    }
    bs += __shfl_xor(bs, 16, 32);
    lsum = lsum * alpha + bs;
    m = mn;
    // ---- rescale accumulators: broadcast alpha for query row (r + 8*hi) ----
    #pragma unroll
    for (int r = 0; r < 8; ++r) {
      float ar = __shfl(alpha, r + hi8, 32);
      #pragma unroll
      for (int t = 0; t < 8; ++t) acc[t][r] *= ar;
    }
    // ---- o += p * v from LDS ----
    #pragma unroll
    for (int t = 0; t < 8; ++t) {
      const __bf16* vp = svc + (t * 16) * 64;
      v16bf b0 = load_b_frag(vp, 64, lane);
      v16bf b1 = load_b_frag(vp + 32, 64, lane);
      acc[t] = wmma_bf16(pA[1], b1, wmma_bf16(pA[0], b0, acc[t]));
    }

    if (has_next) wait_async0();   // drain this wave's async copies
    __syncthreads();
    cur ^= 1;
  }

  // ---- epilogue: out = x + beta * (o / lsum) ----
  float beta = beta_p[0];
  float inv = 1.0f / lsum;
  int col0 = lane & 15, hi = lane >> 4;
  #pragma unroll
  for (int r = 0; r < 8; ++r) {
    float ir = __shfl(inv, r + hi8, 32);
    int tok = qt * 16 + r + 8 * hi;
    size_t rowoff = ((size_t)b * NTOK + tok) * CCH;
    #pragma unroll
    for (int t = 0; t < 8; ++t) {
      int c = cs * 128 + t * 16 + col0;
      out[rowoff + c] = Xin[rowoff + c] + beta * (acc[t][r] * ir);
    }
  }
}

extern "C" void kernel_launch(void* const* d_in, const int* in_sizes, int n_in,
                              void* d_out, int out_size, void* d_ws, size_t ws_size,
                              hipStream_t stream) {
  const float* X    = (const float*)d_in[0];
  const float* Wq   = (const float*)d_in[1];
  const float* bq   = (const float*)d_in[2];
  const float* Wk   = (const float*)d_in[3];
  const float* bk   = (const float*)d_in[4];
  const float* Wv   = (const float*)d_in[5];
  const float* bv   = (const float*)d_in[6];
  const float* beta = (const float*)d_in[7];
  float* out = (float*)d_out;

  char* ws = (char*)d_ws;
  __bf16* qkT = (__bf16*)(ws + QKT_OFF);
  __bf16* wvT = (__bf16*)(ws + WVT_OFF);
  __bf16* qb  = (__bf16*)(ws + Q_OFF);
  __bf16* kb  = (__bf16*)(ws + K_OFF);
  __bf16* vT  = (__bf16*)(ws + VT_OFF);

  // 1) weights -> bf16, transposed (K-contiguous) for B-operand fragments
  wconv_kernel<<<(128 * 512 + 512 * 512 + 255) / 256, 256, 0, stream>>>(Wq, Wk, Wv, qkT, wvT);
  // 2) q,k projection: 256 blocks x 8 waves (128 tokens each, shared weights)
  proj_qk_kernel<<<256, 256, 0, stream>>>(X, qkT, bq, bk, qb, kb);
  // 3) v projection (transposed output): 1024 blocks (4 col-slices x 256 groups)
  proj_v_kernel<<<1024, 256, 0, stream>>>(X, wvT, bv, vT);
  // 4) flash attention: 16 b x 4 cs x 16 query-groups = 1024 blocks of 8 waves
  attn_kernel<<<1024, 256, 0, stream>>>(X, qb, kb, vT, beta, out);
}
